// HNet_88880053223666
// MI455X (gfx1250) — matrix operand order repaired and checked by reading, not compiled, not run on hardware
//
#include <hip/hip_runtime.h>
#include <math.h>

// ---------------------------------------------------------------------------
// HNet forward for MI455X (gfx1250). All GEMMs via v_wmma_f32_16x16x32_f16
// (f32 I/O in memory, f16 compute, f32 accumulate). Working set (~35MB) is
// L2-resident (192MB), so kernels stage whole weight panels in LDS once and
// run fully-unrolled WMMA chains; elementwise/softmax/topk are wave32 VALU.
// ---------------------------------------------------------------------------

typedef __attribute__((ext_vector_type(16))) _Float16 v16h;
typedef __attribute__((ext_vector_type(8)))  float    v8f;

#define BATCH  256
#define SEQ0   256
#define DMODEL 64
#define NHEAD  4
#define VOCABN 50257

__device__ __forceinline__ float gelu_f(float x) {
  float x3 = x * x * x;
  return 0.5f * x * (1.0f + tanhf(0.7978845608028654f * (x + 0.044715f * x3)));
}

__device__ __forceinline__ float wave_sum(float v) {
#pragma unroll
  for (int m = 16; m >= 1; m >>= 1) v += __shfl_xor(v, m, 32);
  return v;
}

// ---------------------------------------------------------------------------
// WMMA GEMM: C[M,N] = act(A[M,K] @ W + (res ? C : 0))
//   Template K in {64, 256} -> fully unrolled k-loop (2 or 8 v_wmma).
//   BT:     W is [N,K] row-major, compute A @ W^T (tied vocab head).
//   BOUNDS: N not a multiple of 64 (gate N=16, vocab N=50257).
//   Block: 128 threads (4 wave32). Tile: 16 rows x 64 cols.
//   grid = (M/16, ceil(N/64)). Whole A (16xK) + W (Kx64) panels staged once.
// ---------------------------------------------------------------------------
template <int K, bool BT, bool BOUNDS>
__global__ void gemm_wmma_kernel(const float* __restrict__ A,
                                 const float* __restrict__ W,
                                 float* __restrict__ C,
                                 int N, int lda, int act, int res) {
  constexpr int AS = K + 12;  // row stride: 16B-aligned, banks spread (mod64=12)
  constexpr int WS = 68;      // row stride: 16B-aligned, banks spread (mod64=4)
  __shared__ __align__(16) float Als[16 * AS];
  __shared__ __align__(16) float Wls[K * WS];

  const int tid  = threadIdx.x;
  const int wave = tid >> 5;
  const int lane = tid & 31;
  const int m0 = blockIdx.x * 16;
  const int n0 = blockIdx.y * 64;

  // Stage A: 16 x K, float4 loads (rows 16B-aligned: lda in {64,256,16384}).
#pragma unroll
  for (int i = tid; i < 4 * K; i += 128) {          // 16*(K/4) float4s
    int r = i / (K / 4);
    int c = (i - r * (K / 4)) * 4;
    const float4 v = *(const float4*)(A + (size_t)(m0 + r) * lda + c);
    float4* dst = (float4*)&Als[r * AS + c];
    *dst = v;
  }

  // Stage W: K x 64 panel.
  if (!BT) {
#pragma unroll
    for (int i = tid; i < K * 16; i += 128) {       // K rows x 16 float4s
      int kk = i >> 4;
      int c  = (i & 15) * 4;
      if (!BOUNDS) {
        *(float4*)&Wls[kk * WS + c] = *(const float4*)(W + (size_t)kk * N + n0 + c);
      } else {
#pragma unroll
        for (int u = 0; u < 4; u++) {
          int n = n0 + c + u;
          Wls[kk * WS + c + u] = (n < N) ? W[(size_t)kk * N + n] : 0.0f;
        }
      }
    }
  } else {
    // W is [N,K]: read float4 along K (contiguous), scatter-transpose to LDS.
#pragma unroll
    for (int i = tid; i < 16 * K; i += 128) {       // 64 n x (K/4) float4s
      int nn = i / (K / 4);
      int c  = (i - nn * (K / 4)) * 4;
      int n  = n0 + nn;
      float4 v = make_float4(0.0f, 0.0f, 0.0f, 0.0f);
      if (!BOUNDS || n < N) v = *(const float4*)(W + (size_t)n * K + c);
      Wls[(c + 0) * WS + nn] = v.x;
      Wls[(c + 1) * WS + nn] = v.y;
      Wls[(c + 2) * WS + nn] = v.z;
      Wls[(c + 3) * WS + nn] = v.w;
    }
  }
  __syncthreads();

  const int mrow = lane & 15;   // A fragment: m = lane%16
  const int half = lane >> 4;   // A fragment: K-half select
  const int nl   = lane & 15;   // B fragment: n = lane%16
  const int kg   = lane >> 4;   // B fragment: K-group select

  v8f acc = {};
#pragma unroll
  for (int k0 = 0; k0 < K; k0 += 32) {
    // Fragments in contiguous k order, k = 16*(lane/16) + e for A and B (f32 -> f16 convert on read).
    v16h a, b;
#pragma unroll
    for (int e = 0; e < 16; e++) {
      int ka = k0 + (half << 4) + e;                          // A 16x32 layout
      a[e] = (_Float16)Als[mrow * AS + ka];
      int kb = k0 + (kg << 4) + e;                            // B 32x16 layout
      b[e] = (_Float16)Wls[kb * WS + (wave << 4) + nl];
    }
    acc = __builtin_amdgcn_wmma_f32_16x16x32_f16(false, a, false, b,
                                                 (short)0, acc, false, false);
  }

  // C/D layout: lane holds n = lane%16; VGPR r -> m = r + 8*(lane/16).
  int n = n0 + (wave << 4) + nl;
  if (!BOUNDS || n < N) {
#pragma unroll
    for (int r = 0; r < 8; r++) {
      size_t ci = (size_t)(m0 + r + (half << 3)) * (size_t)N + n;
      float v = acc[r];
      if (res) v += C[ci];
      if (act) v = gelu_f(v);
      C[ci] = v;
    }
  }
}

// ---------------------------------------------------------------------------
// Elementwise / reduction kernels (rows of 64, one wave per row)
// ---------------------------------------------------------------------------
__global__ void embed_rms_kernel(const int* __restrict__ idx,
                                 const float* __restrict__ tok,
                                 const float* __restrict__ pos,
                                 float* __restrict__ x, int T) {
  size_t row = blockIdx.x;
  int l = threadIdx.x;
  int t = (int)(row % (size_t)T);
  int ix = idx[row];
  float a = tok[(size_t)ix * 64 + l]      + pos[t * 64 + l];
  float b = tok[(size_t)ix * 64 + 32 + l] + pos[t * 64 + 32 + l];
  float ss = wave_sum(a * a + b * b);
  float r = rsqrtf(ss * (1.0f / 64.0f) + 1e-8f);
  x[row * 64 + l]      = a * r;
  x[row * 64 + 32 + l] = b * r;
}

__global__ void ln64_kernel(const float* __restrict__ x,
                            const float* __restrict__ w,
                            float* __restrict__ y) {
  size_t row = blockIdx.x;
  int l = threadIdx.x;
  float a = x[row * 64 + l], b = x[row * 64 + 32 + l];
  float mu = wave_sum(a + b) * (1.0f / 64.0f);
  float da = a - mu, db = b - mu;
  float var = wave_sum(da * da + db * db) * (1.0f / 64.0f);
  float r = rsqrtf(var + 1e-5f);
  y[row * 64 + l]      = da * r * w[l];
  y[row * 64 + 32 + l] = db * r * w[l + 32];
}

__global__ void rms64_kernel(float* __restrict__ x) {
  size_t row = blockIdx.x;
  int l = threadIdx.x;
  float a = x[row * 64 + l], b = x[row * 64 + 32 + l];
  float ss = wave_sum(a * a + b * b);
  float r = rsqrtf(ss * (1.0f / 64.0f) + 1e-8f);
  x[row * 64 + l]      = a * r;
  x[row * 64 + 32 + l] = b * r;
}

// LayerNorm over rows of length N (gate rows, N = S). blockDim = N (<=256).
__global__ void lnrow_kernel(const float* __restrict__ x,
                             const float* __restrict__ w,
                             float* __restrict__ y, int N) {
  __shared__ float red[256];
  int row = blockIdx.x, i = threadIdx.x;
  float v = x[(size_t)row * N + i];
  red[i] = v;
  __syncthreads();
  for (int s = blockDim.x >> 1; s > 0; s >>= 1) {
    if (i < s) red[i] += red[i + s];
    __syncthreads();
  }
  float mu = red[0] / (float)N;
  __syncthreads();
  float d = v - mu;
  red[i] = d * d;
  __syncthreads();
  for (int s = blockDim.x >> 1; s > 0; s >>= 1) {
    if (i < s) red[i] += red[i + s];
    __syncthreads();
  }
  float var = red[0] / (float)N;
  y[(size_t)row * N + i] = d * rsqrtf(var + 1e-5f) * w[i];
}

// ---------------------------------------------------------------------------
// Causal MHA core: dh=16, grid=(B, NH), blockDim=S. K/V head staged in LDS.
// Matches reference softmax: masked logits are -1e9 -> exp underflows to 0.
// ---------------------------------------------------------------------------
__global__ void attn_kernel(const float* __restrict__ Qm,
                            const float* __restrict__ Km,
                            const float* __restrict__ Vm,
                            float* __restrict__ Om, int S) {
  __shared__ float kls[256 * 16];
  __shared__ float vls[256 * 16];
  int b = blockIdx.x, h = blockIdx.y, t = threadIdx.x;
  size_t base = ((size_t)b * S) * 64 + h * 16;
#pragma unroll
  for (int d = 0; d < 16; d++) {
    kls[t * 16 + d] = Km[base + (size_t)t * 64 + d];
    vls[t * 16 + d] = Vm[base + (size_t)t * 64 + d];
  }
  __syncthreads();
  float qv[16];
#pragma unroll
  for (int d = 0; d < 16; d++) qv[d] = Qm[base + (size_t)t * 64 + d];

  float mx = -1e30f;
  for (int j = 0; j <= t; j++) {
    float s = 0.0f;
#pragma unroll
    for (int d = 0; d < 16; d++) s += qv[d] * kls[j * 16 + d];
    mx = fmaxf(mx, s * 0.25f);
  }
  float sum = 0.0f;
  float acc[16];
#pragma unroll
  for (int d = 0; d < 16; d++) acc[d] = 0.0f;
  for (int j = 0; j <= t; j++) {
    float s = 0.0f;
#pragma unroll
    for (int d = 0; d < 16; d++) s += qv[d] * kls[j * 16 + d];
    float e = expf(s * 0.25f - mx);
    sum += e;
#pragma unroll
    for (int d = 0; d < 16; d++) acc[d] += e * vls[j * 16 + d];
  }
  float inv = 1.0f / sum;
#pragma unroll
  for (int d = 0; d < 16; d++) Om[base + (size_t)t * 64 + d] = acc[d] * inv;
}

// gate[row] = dot(gact[row,0:16], g2) + gb2
__global__ void gate_reduce_kernel(const float* __restrict__ gact,
                                   const float* __restrict__ g2,
                                   const float* __restrict__ gb2,
                                   float* __restrict__ gate, int Mtot) {
  int r = blockIdx.x * blockDim.x + threadIdx.x;
  if (r < Mtot) {
    float s = gb2[0];
#pragma unroll
    for (int e = 0; e < 16; e++) s += gact[(size_t)r * 16 + e] * g2[e];
    gate[r] = s;
  }
}

// top-k with jax.lax.top_k tie-breaking, output indices ascending.
__global__ void topk_kernel(const float* __restrict__ gate,
                            int* __restrict__ idx, int S, int KSEL) {
  __shared__ float gl[256];
  __shared__ int sel[256];
  int b = blockIdx.x, i = threadIdx.x;
  gl[i] = gate[(size_t)b * S + i];
  __syncthreads();
  float gi = gl[i];
  int cnt = 0;
  for (int j = 0; j < S; j++)
    cnt += (gl[j] > gi) || (gl[j] == gi && j < i);
  sel[i] = (cnt < KSEL) ? 1 : 0;
  __syncthreads();
  if (sel[i]) {
    int r = 0;
    for (int j = 0; j < i; j++) r += sel[j];
    idx[b * KSEL + r] = i;
  }
}

// out[b,j,:] = g[b, idx[b,j], :] * sigmoid(gate[b, idx[b,j]])
__global__ void gather_kernel(const float* __restrict__ g,
                              const float* __restrict__ gate,
                              const int* __restrict__ idx,
                              float* __restrict__ out, int S, int KSEL) {
  int blk = blockIdx.x;
  int b = blk / KSEL, j = blk - b * KSEL;
  int src = idx[b * KSEL + j];
  float gv = gate[(size_t)b * S + src];
  float sg = 1.0f / (1.0f + expf(-gv));
  out[(size_t)blk * 64 + threadIdx.x] =
      g[((size_t)b * S + src) * 64 + threadIdx.x] * sg;
}

__global__ void fillpos_kernel(float* __restrict__ out,
                               const float* __restrict__ pos, int S) {
  size_t row = blockIdx.x;
  out[row * 64 + threadIdx.x] = pos[(row % (size_t)S) * 64 + threadIdx.x];
}

// out[b, idx[b,j], :] = x[b,j,:] + pos[idx[b,j], :]
__global__ void scatter_kernel(float* __restrict__ out,
                               const float* __restrict__ x,
                               const int* __restrict__ idx,
                               const float* __restrict__ pos,
                               int S, int KSEL) {
  int blk = blockIdx.x;
  int b = blk / KSEL, j = blk - b * KSEL;
  int t = idx[b * KSEL + j];
  int c = threadIdx.x;
  out[((size_t)b * S + t) * 64 + c] =
      x[(size_t)blk * 64 + c] + pos[(size_t)t * 64 + c];
}

__global__ void addpos_kernel(const float* __restrict__ xres,
                              const float* __restrict__ pos,
                              float* __restrict__ out, int S) {
  size_t row = blockIdx.x;
  int c = threadIdx.x;
  out[row * 64 + c] = xres[row * 64 + c] + pos[(row % (size_t)S) * 64 + c];
}

// ---------------------------------------------------------------------------
// Host-side orchestration
// ---------------------------------------------------------------------------
struct BlkP  { const float *ln1,*wq,*wk,*wv,*wo,*ln2,*w1,*w2; };
struct CombP { const float *lnq,*lnx,*wq,*wk,*wv,*wo,*ln2,*w1,*w2; };
struct PatchP {
  BlkP query_block;
  const float *g1,*g2,*gb2,*gate_norm,*up_proj;
  BlkP proj_block;
  const float *down_norm,*out_pos_emb;
  BlkP down_block;
  const float *down_proj,*res_norm;
  CombP down_scatter2;
  BlkP down_scan;
};

struct Ctx {
  hipStream_t st;
  float *X, *RES1, *XC, *G, *X2, *SC, *T, *H;   // big buffers
  float *Q, *K, *V, *O, *GACT;                  // aliases into H
  float *GATE0, *GATE1;
  int *IDX0, *IDX1;
};

template <int K, bool BT, bool BOUNDS>
static void gemm(Ctx& c, const float* A, const float* W, float* C,
                 int M, int N, int lda, int act, int res) {
  dim3 g(M / 16, (N + 63) / 64);
  gemm_wmma_kernel<K, BT, BOUNDS><<<g, 128, 0, c.st>>>(A, W, C, N, lda, act, res);
}

static void ln64(Ctx& c, const float* x, const float* w, float* y, int M) {
  ln64_kernel<<<M, 32, 0, c.st>>>(x, w, y);
}

// x = x + mha(ln1(x)); x = x + mlp(ln2(x));  (in-place on x)
static void run_block(Ctx& c, float* x, int M, int S, const BlkP& p) {
  ln64(c, x, p.ln1, c.T, M);
  gemm<64, false, false>(c, c.T, p.wq, c.Q, M, 64, 64, 0, 0);
  gemm<64, false, false>(c, c.T, p.wk, c.K, M, 64, 64, 0, 0);
  gemm<64, false, false>(c, c.T, p.wv, c.V, M, 64, 64, 0, 0);
  attn_kernel<<<dim3(BATCH, NHEAD), S, 0, c.st>>>(c.Q, c.K, c.V, c.O, S);
  gemm<64, false, false>(c, c.O, p.wo, x, M, 64, 64, 0, 1);
  ln64(c, x, p.ln2, c.T, M);
  gemm<64, false, false>(c, c.T, p.w1, c.H, M, 256, 64, 1, 0);   // gelu
  gemm<256, false, false>(c, c.H, p.w2, x, M, 64, 256, 0, 1);
}

// qb = qb + mha(ln(qb,lnq), ln(kv,lnx)); qb = qb + mlp(ln(qb,ln2))
static void run_combine(Ctx& c, const float* kv, float* qb, int M, int S,
                        const CombP& p) {
  ln64(c, qb, p.lnq, c.T, M);
  ln64(c, kv, p.lnx, c.X2, M);
  gemm<64, false, false>(c, c.T,  p.wq, c.Q, M, 64, 64, 0, 0);
  gemm<64, false, false>(c, c.X2, p.wk, c.K, M, 64, 64, 0, 0);
  gemm<64, false, false>(c, c.X2, p.wv, c.V, M, 64, 64, 0, 0);
  attn_kernel<<<dim3(BATCH, NHEAD), S, 0, c.st>>>(c.Q, c.K, c.V, c.O, S);
  gemm<64, false, false>(c, c.O, p.wo, qb, M, 64, 64, 0, 1);
  ln64(c, qb, p.ln2, c.T, M);
  gemm<64, false, false>(c, c.T, p.w1, c.H, M, 256, 64, 1, 0);
  gemm<256, false, false>(c, c.H, p.w2, qb, M, 64, 256, 0, 1);
}

// abstract_up: input x preserved by caller as residual; G holds a copy.
static void run_up(Ctx& c, const PatchP& p, int Sout, int KSEL,
                   float* gate, int* idxb) {
  int Mout = BATCH * Sout;
  int Min  = BATCH * KSEL;
  run_block(c, c.G, Mout, Sout, p.query_block);
  gemm<64, false, true>(c, c.G, p.g1, c.GACT, Mout, 16, 64, 1, 0); // gelu(g@g1)
  gate_reduce_kernel<<<Mout / 256, 256, 0, c.st>>>(c.GACT, p.g2, p.gb2, gate, Mout);
  lnrow_kernel<<<BATCH, Sout, 0, c.st>>>(gate, p.gate_norm, gate, Sout);
  rms64_kernel<<<Mout, 32, 0, c.st>>>(c.G);
  topk_kernel<<<BATCH, Sout, 0, c.st>>>(gate, idxb, Sout, KSEL);
  gather_kernel<<<BATCH * KSEL, 64, 0, c.st>>>(c.G, gate, idxb, c.X2, Sout, KSEL);
  gemm<64, false, false>(c, c.X2, p.up_proj, c.XC, Min, 64, 64, 0, 0);
  run_block(c, c.XC, Min, KSEL, p.proj_block);
}

// abstract_down: consumes c.XC (len Sin), residual res (len Sout),
// indices idxb; leaves result in c.XC (len Sout).
static void run_down(Ctx& c, const float* res, const int* idxb,
                     const PatchP& p, int Sin, int Sout) {
  int Min = BATCH * Sin, Mout = BATCH * Sout;
  ln64(c, c.XC, p.down_norm, c.X2, Min);                               // ln(x)
  fillpos_kernel<<<Mout, 64, 0, c.st>>>(c.SC, p.out_pos_emb, Sout);    // pos bg
  scatter_kernel<<<Min, 64, 0, c.st>>>(c.SC, c.X2, idxb, p.out_pos_emb,
                                       Sout, Sin);                     // x+pos
  run_block(c, c.SC, Mout, Sout, p.down_block);
  gemm<64, false, false>(c, c.SC, p.down_proj, c.G, Mout, 64, 64, 0, 0);
  addpos_kernel<<<Mout, 64, 0, c.st>>>(res, p.out_pos_emb, c.X2, Sout);
  ln64(c, c.X2, p.res_norm, c.SC, Mout);                               // query
  run_combine(c, c.G, c.SC, Mout, Sout, p.down_scatter2);
  run_block(c, c.SC, Mout, Sout, p.down_scan);
  hipMemcpyAsync(c.XC, c.SC, (size_t)Mout * 64 * sizeof(float),
                 hipMemcpyDeviceToDevice, c.st);
}

extern "C" void kernel_launch(void* const* d_in, const int* in_sizes, int n_in,
                              void* d_out, int out_size, void* d_ws, size_t ws_size,
                              hipStream_t stream) {
  (void)in_sizes; (void)n_in; (void)out_size; (void)ws_size;

  // --- parse inputs in setup_inputs() dict-insertion order ---
  const int* idx = (const int*)d_in[0];
  int pi = 1;
  auto nf = [&]() { return (const float*)d_in[pi++]; };
  auto rdBlk = [&]() {
    BlkP p; p.ln1 = nf(); p.wq = nf(); p.wk = nf(); p.wv = nf();
    p.wo = nf(); p.ln2 = nf(); p.w1 = nf(); p.w2 = nf(); return p;
  };
  auto rdComb = [&]() {
    CombP p; p.lnq = nf(); p.lnx = nf(); p.wq = nf(); p.wk = nf(); p.wv = nf();
    p.wo = nf(); p.ln2 = nf(); p.w1 = nf(); p.w2 = nf(); return p;
  };
  auto rdPatch = [&]() {
    PatchP p;
    p.query_block = rdBlk();
    p.g1 = nf(); p.g2 = nf(); p.gb2 = nf(); p.gate_norm = nf(); p.up_proj = nf();
    p.proj_block = rdBlk();
    p.down_norm = nf(); p.out_pos_emb = nf();
    p.down_block = rdBlk();
    p.down_proj = nf(); p.res_norm = nf();
    p.down_scatter2 = rdComb();
    p.down_scan = rdBlk();
    return p;
  };
  const float* tok_emb = nf();
  const float* pos_emb = nf();
  const float* h_ln    = nf();
  PatchP patch0 = rdPatch();
  PatchP patch1 = rdPatch();
  BlkP inner[4];
  for (int i = 0; i < 4; i++) inner[i] = rdBlk();

  // --- workspace layout (floats); peak ~185MB ---
  const size_t M64 = (size_t)65536 * 64;   // 16MB per slot
  float* ws = (float*)d_ws;
  Ctx c;
  c.st   = stream;
  c.X    = ws + 0 * M64;     // residual of patch0 (embedding output)
  c.RES1 = ws + 1 * M64;     // residual of patch1
  c.XC   = ws + 2 * M64;     // current activation
  c.G    = ws + 3 * M64;     // query-block output / pds
  c.X2   = ws + 4 * M64;     // gather buffer / ln temps
  c.SC   = ws + 5 * M64;     // scattered / combine in-out
  c.T    = ws + 6 * M64;     // ln output feeding GEMMs
  c.H    = ws + 7 * M64;     // 4 slots: MLP hidden (M x 256)
  c.Q    = c.H;              // Q/K/V/O alias H: live ranges don't overlap
  c.K    = c.H + 1 * M64;
  c.V    = c.H + 2 * M64;
  c.O    = c.H + 3 * M64;
  c.GACT = c.H;              // gate MLP hidden (M x 16), H free at that point
  c.GATE0 = ws + 11 * M64;
  c.GATE1 = c.GATE0 + 65536;
  c.IDX0  = (int*)(c.GATE1 + 32768);
  c.IDX1  = c.IDX0 + BATCH * 128;

  const size_t rowBytes = (size_t)64 * sizeof(float);

  // --- forward pass ---
  // x = rms(tok_emb[idx] + pos)  -> X (kept as patch0 residual)
  embed_rms_kernel<<<BATCH * SEQ0, 32, 0, stream>>>(idx, tok_emb, pos_emb, c.X, SEQ0);

  // patch0 abstract_up: g = block(copy of x); gate; gather 256 -> 128
  hipMemcpyAsync(c.G, c.X, (size_t)BATCH * SEQ0 * rowBytes,
                 hipMemcpyDeviceToDevice, stream);
  run_up(c, patch0, /*Sout=*/256, /*KSEL=*/128, c.GATE0, c.IDX0);

  // patch1 abstract_up: 128 -> 64
  hipMemcpyAsync(c.RES1, c.XC, (size_t)BATCH * 128 * rowBytes,
                 hipMemcpyDeviceToDevice, stream);
  hipMemcpyAsync(c.G, c.XC, (size_t)BATCH * 128 * rowBytes,
                 hipMemcpyDeviceToDevice, stream);
  run_up(c, patch1, /*Sout=*/128, /*KSEL=*/64, c.GATE1, c.IDX1);

  // inner blocks at S=64
  for (int i = 0; i < 4; i++) run_block(c, c.XC, BATCH * 64, 64, inner[i]);

  // abstract_down patch1 (64 -> 128), then patch0 (128 -> 256)
  run_down(c, c.RES1, c.IDX1, patch1, 64, 128);
  run_down(c, c.X,    c.IDX0, patch0, 128, 256);

  // final LN + tied-head logits for last token: [256,64] @ tok_emb^T
  ln64(c, c.XC, h_ln, c.T, BATCH * SEQ0);
  gemm<64, true, true>(c, c.T + (size_t)(SEQ0 - 1) * 64, tok_emb, (float*)d_out,
                       BATCH, VOCABN, /*lda=*/SEQ0 * 64, /*act=*/0, /*res=*/0);
}
